// GraphAttention_86835648790655
// MI455X (gfx1250) — matrix operand order repaired and checked by reading, not compiled, not run on hardware
//
#include <hip/hip_runtime.h>

#define N_NODES   50000
#define N_EDGES   800000
#define F_IN      256
#define F_OUT     64
#define HEADS     4
#define ALPHA_LK  0.2f

typedef __attribute__((ext_vector_type(4)))  float        f32x4;
typedef __attribute__((ext_vector_type(8)))  float        v8f;
typedef __attribute__((ext_vector_type(8)))  _Float16     h8;
typedef __attribute__((ext_vector_type(16))) _Float16     v16h;
typedef __attribute__((ext_vector_type(4)))  unsigned int u32x4;
typedef __attribute__((ext_vector_type(4)))  int          i32x4;
typedef __attribute__((ext_vector_type(8)))  int          i32x8;

// ---------------------------------------------------------------------------
// LDS layout for the GEMM kernel (dynamic shared memory)
//   lA : 128 x 260 f32  (TDM-loaded A strip, 16B-pad per 256-dword row)
//   lW : 256 x 64  f32  (TDM-loaded W[h], raw)
//   sB : 64 x 264  f16  (converted, transposed [n][k])
// ---------------------------------------------------------------------------
#define LDA_STR   260                           // floats per A row in LDS
#define LDB_STR   264                           // halves per B row in LDS
#define OFF_LW    (128 * LDA_STR * 4)           // 133120
#define OFF_SB    (OFF_LW + F_IN * F_OUT * 4)   // 198656
#define SMEM_TOT  (OFF_SB + F_OUT * LDB_STR * 2) // 232448 bytes (< 320KB WGP)

// 2-D tensor_load_to_lds: tile (tile0 x tile1) of f32, row stride = stride0
__device__ __forceinline__ void tdm_load_2d(unsigned long long gaddr,
                                            unsigned int lds_addr,
                                            unsigned int dim0, unsigned int dim1,
                                            unsigned int tile0, unsigned int tile1,
                                            unsigned int stride0, bool pad)
{
    u32x4 g0;
    g0.x = 1u;                                              // count=1, user mode
    g0.y = lds_addr;                                        // LDS byte address
    g0.z = (unsigned int)(gaddr & 0xFFFFFFFFull);           // global_addr[31:0]
    g0.w = (unsigned int)((gaddr >> 32) & 0x1FFFFFFull)     // global_addr[56:32]
         | (2u << 30);                                      // type = 2 ("image")

    unsigned int flags = (2u << 16);                        // data_size = 4 bytes
    if (pad)                                                // pad 4 dwords per 256 dwords
        flags |= (1u << 20) | (7u << 22) | (3u << 25);
    i32x8 g1;
    g1[0] = (int)flags;
    g1[1] = (int)((dim0 & 0xFFFFu) << 16);                  // tensor_dim0[15:0]
    g1[2] = (int)((dim0 >> 16) | ((dim1 & 0xFFFFu) << 16)); // dim0[31:16] | dim1[15:0]
    g1[3] = (int)((dim1 >> 16) | (tile0 << 16));            // dim1[31:16] | tile_dim0
    g1[4] = (int)tile1;                                     // tile_dim1 (tile_dim2 = 0)
    g1[5] = (int)stride0;                                   // tensor_dim0_stride[31:0]
    g1[6] = 0;
    g1[7] = 0;

    i32x4 gz = {0, 0, 0, 0};
#if defined(__clang_major__) && (__clang_major__ >= 23)
    i32x8 gz8 = {0, 0, 0, 0, 0, 0, 0, 0};
    __builtin_amdgcn_tensor_load_to_lds(g0, g1, gz, gz, gz8, 0);
#else
    __builtin_amdgcn_tensor_load_to_lds(g0, g1, gz, gz, 0);
#endif
}

// ---------------------------------------------------------------------------
// Kernel 1: feats[h] = x @ W[h] via TDM staging + v_wmma_f32_16x16x32_f16
// block = 256 threads (8 waves); tile = 128 rows x 64 cols; K unrolled (32 wmma)
// ---------------------------------------------------------------------------
__global__ __launch_bounds__(256)
void gat_proj_wmma_kernel(const float* __restrict__ x,
                          const float* __restrict__ W,
                          float* __restrict__ feats)
{
    extern __shared__ char smem[];
    float*    lA = (float*)smem;
    float*    lW = (float*)(smem + OFF_LW);
    _Float16* sB = (_Float16*)(smem + OFF_SB);

    const int tid    = threadIdx.x;
    const int wave   = tid >> 5;
    const int lane   = tid & 31;
    const int lane16 = lane & 15;
    const int hsel   = lane >> 4;
    const int h      = blockIdx.y;
    const int mBase  = blockIdx.x * 128;

    // ---- phase 1: TDM bulk loads (one wave issues; EXEC-masked waves skip) ----
    if (tid == 0) {
        // A strip: rows [mBase, mBase+128) x 256; OOB rows (tail block) read 0
        tdm_load_2d((unsigned long long)(uintptr_t)(x + (size_t)mBase * F_IN),
                    (unsigned int)(uintptr_t)lA,
                    F_IN, (unsigned int)(N_NODES - mBase),
                    F_IN, 128, F_IN, /*pad=*/true);
        // W[h]: 256 x 64 f32, contiguous
        tdm_load_2d((unsigned long long)(uintptr_t)(W + (size_t)h * F_IN * F_OUT),
                    (unsigned int)(uintptr_t)lW,
                    F_OUT, F_IN, F_OUT, F_IN, F_OUT, /*pad=*/false);
        __builtin_amdgcn_s_wait_tensorcnt(0);
    }
    __syncthreads();

    // ---- phase 2: convert W f32 [k][n] -> f16 transposed sB[n][k] ----
    {
        const int k = tid;                       // 0..255
        const f32x4* wr = (const f32x4*)(lW + (size_t)k * F_OUT);
        #pragma unroll
        for (int j = 0; j < 16; ++j) {
            f32x4 v = wr[j];
            const int n = j * 4;
            sB[(n + 0) * LDB_STR + k] = (_Float16)v.x;
            sB[(n + 1) * LDB_STR + k] = (_Float16)v.y;
            sB[(n + 2) * LDB_STR + k] = (_Float16)v.z;
            sB[(n + 3) * LDB_STR + k] = (_Float16)v.w;
        }
    }
    __syncthreads();

    // ---- phase 3: pure LDS + WMMA inner loop (no globals, no barriers) ----
    v8f acc[4];
    #pragma unroll
    for (int t = 0; t < 4; ++t) acc[t] = (v8f){0,0,0,0,0,0,0,0};

    const float* aRow = lA + (size_t)(wave * 16 + lane16) * LDA_STR;

    #pragma unroll
    for (int k0 = 0; k0 < F_IN; k0 += 32) {
        // A frag 16x32: lanes 0-15 K={0..7,16..23}, lanes 16-31 K={8..15,24..31}
        f32x4 a0 = *(const f32x4*)(aRow + k0 + 8 * hsel);
        f32x4 a1 = *(const f32x4*)(aRow + k0 + 8 * hsel + 4);
        f32x4 a2 = *(const f32x4*)(aRow + k0 + 16 + 8 * hsel);
        f32x4 a3 = *(const f32x4*)(aRow + k0 + 16 + 8 * hsel + 4);
        v16h afrag = {(_Float16)a0.x,(_Float16)a0.y,(_Float16)a0.z,(_Float16)a0.w,
                      (_Float16)a1.x,(_Float16)a1.y,(_Float16)a1.z,(_Float16)a1.w,
                      (_Float16)a2.x,(_Float16)a2.y,(_Float16)a2.z,(_Float16)a2.w,
                      (_Float16)a3.x,(_Float16)a3.y,(_Float16)a3.z,(_Float16)a3.w};

        #pragma unroll
        for (int t = 0; t < 4; ++t) {
            // B frag 32x16: lanes 0-15 K={0..7,16..23}, lanes 16-31 K={8..15,24..31}
            const _Float16* bp = sB + (size_t)(t * 16 + lane16) * LDB_STR
                                    + k0 + 8 * hsel;
            h8 b0 = *(const h8*)(bp);
            h8 b1 = *(const h8*)(bp + 16);
            v16h bfrag = __builtin_shufflevector(b0, b1,
                           0,1,2,3,4,5,6,7,8,9,10,11,12,13,14,15);
            acc[t] = __builtin_amdgcn_wmma_f32_16x16x32_f16(
                         false, afrag, false, bfrag,
                         (short)0, acc[t], false, false);
        }
    }

    // ---- epilogue: uniform fast path for full tiles ----
    float* outH = feats + (size_t)h * N_NODES * F_OUT;
    const int rBase = mBase + wave * 16 + 8 * hsel;
    if (mBase + 128 <= N_NODES) {
        #pragma unroll
        for (int t = 0; t < 4; ++t)
            #pragma unroll
            for (int r = 0; r < 8; ++r)
                outH[(size_t)(rBase + r) * F_OUT + t * 16 + lane16] = acc[t][r];
    } else {
        #pragma unroll
        for (int t = 0; t < 4; ++t)
            #pragma unroll
            for (int r = 0; r < 8; ++r)
                if (rBase + r < N_NODES)
                    outH[(size_t)(rBase + r) * F_OUT + t * 16 + lane16] = acc[t][r];
    }
}

// ---------------------------------------------------------------------------
// Kernel 2: attention score dots, one wave per (h, node)
// ---------------------------------------------------------------------------
__global__ __launch_bounds__(256)
void gat_scores_kernel(const float* __restrict__ feats,
                       const float* __restrict__ a_self,
                       const float* __restrict__ a_neigh,
                       float* __restrict__ s_self,
                       float* __restrict__ s_neigh)
{
    const int wid = blockIdx.x * 8 + (threadIdx.x >> 5);   // = h*N + n
    if (wid >= HEADS * N_NODES) return;
    const int lane = threadIdx.x & 31;
    const int h = wid / N_NODES;

    float2 f  = *(const float2*)(feats   + (size_t)wid * F_OUT + lane * 2);
    float2 as = *(const float2*)(a_self  + h * F_OUT + lane * 2);
    float2 an = *(const float2*)(a_neigh + h * F_OUT + lane * 2);
    float vs = f.x * as.x + f.y * as.y;
    float vn = f.x * an.x + f.y * an.y;
    #pragma unroll
    for (int off = 16; off > 0; off >>= 1) {
        vs += __shfl_xor(vs, off, 32);
        vn += __shfl_xor(vn, off, 32);
    }
    if (lane == 0) { s_self[wid] = vs; s_neigh[wid] = vn; }
}

// ---------------------------------------------------------------------------
// Kernel 3: init out=0, m=-inf, ssum=0
// ---------------------------------------------------------------------------
__global__ void gat_init_kernel(float* __restrict__ out,
                                float* __restrict__ m,
                                float* __restrict__ ssum)
{
    const int i = blockIdx.x * blockDim.x + threadIdx.x;
    if (i < N_NODES * (HEADS * F_OUT)) out[i] = 0.0f;
    if (i < HEADS * N_NODES) {
        m[i]    = -__builtin_inff();
        ssum[i] = 0.0f;
    }
}

__device__ __forceinline__ float leaky_edge(const float* s_self, const float* s_neigh,
                                            const float* adj_val, int h, int row, int col, int e)
{
    float z = s_self[h * N_NODES + row] + s_neigh[h * N_NODES + col];
    float v = (z > 0.0f) ? z : ALPHA_LK * z;
    return v * adj_val[e];
}

__device__ __forceinline__ void atomicMaxFloat(float* addr, float value)
{
    if (value >= 0.0f)
        atomicMax((int*)addr, __float_as_int(value));
    else
        atomicMin((unsigned int*)addr, __float_as_uint(value));
}

// ---------------------------------------------------------------------------
// Kernel 4: segment max over edges (per head)
// ---------------------------------------------------------------------------
__global__ void gat_segmax_kernel(const float* __restrict__ s_self,
                                  const float* __restrict__ s_neigh,
                                  const float* __restrict__ adj_val,
                                  const int* __restrict__ edge_row,
                                  const int* __restrict__ edge_col,
                                  float* __restrict__ m)
{
    const int i = blockIdx.x * blockDim.x + threadIdx.x;
    if (i >= HEADS * N_EDGES) return;
    const int h = i / N_EDGES;
    const int e = i - h * N_EDGES;
    const int row = edge_row[e], col = edge_col[e];
    float ev = leaky_edge(s_self, s_neigh, adj_val, h, row, col, e);
    atomicMaxFloat(&m[h * N_NODES + row], ev);
}

// ---------------------------------------------------------------------------
// Kernel 5: segment sum of exp(e - m)
// ---------------------------------------------------------------------------
__global__ void gat_segsum_kernel(const float* __restrict__ s_self,
                                  const float* __restrict__ s_neigh,
                                  const float* __restrict__ adj_val,
                                  const int* __restrict__ edge_row,
                                  const int* __restrict__ edge_col,
                                  const float* __restrict__ m,
                                  float* __restrict__ ssum)
{
    const int i = blockIdx.x * blockDim.x + threadIdx.x;
    if (i >= HEADS * N_EDGES) return;
    const int h = i / N_EDGES;
    const int e = i - h * N_EDGES;
    const int row = edge_row[e], col = edge_col[e];
    float ev = leaky_edge(s_self, s_neigh, adj_val, h, row, col, e);
    float ex = __expf(ev - m[h * N_NODES + row]);
    atomicAdd(&ssum[h * N_NODES + row], ex);
}

// ---------------------------------------------------------------------------
// Kernel 6: aggregation scatter: out[row, h*64+o] += alpha * feats[h][col][o]
// ---------------------------------------------------------------------------
__global__ __launch_bounds__(256)
void gat_aggregate_kernel(const float* __restrict__ feats,
                          const float* __restrict__ s_self,
                          const float* __restrict__ s_neigh,
                          const float* __restrict__ adj_val,
                          const int* __restrict__ edge_row,
                          const int* __restrict__ edge_col,
                          const float* __restrict__ m,
                          const float* __restrict__ ssum,
                          float* __restrict__ out)
{
    const int e   = blockIdx.x;
    const int tid = threadIdx.x;
    const int h   = tid >> 6;
    const int o   = tid & 63;
    const int row = edge_row[e], col = edge_col[e];

    float ev = leaky_edge(s_self, s_neigh, adj_val, h, row, col, e);
    float w  = __expf(ev - m[h * N_NODES + row]) / ssum[h * N_NODES + row];
    float v  = w * feats[((size_t)h * N_NODES + col) * F_OUT + o];
    atomicAdd(out + (size_t)row * (HEADS * F_OUT) + tid, v);
}

// ---------------------------------------------------------------------------
// Kernel 7: bias + relu epilogue
// ---------------------------------------------------------------------------
__global__ void gat_bias_relu_kernel(float* __restrict__ out,
                                     const float* __restrict__ bias)
{
    const int i = blockIdx.x * blockDim.x + threadIdx.x;
    if (i >= N_NODES * HEADS * F_OUT) return;
    const int c = i & (HEADS * F_OUT - 1);
    float v = out[i] + bias[c];
    out[i] = v > 0.0f ? v : 0.0f;
}

// ---------------------------------------------------------------------------
extern "C" void kernel_launch(void* const* d_in, const int* in_sizes, int n_in,
                              void* d_out, int out_size, void* d_ws, size_t ws_size,
                              hipStream_t stream)
{
    const float* x        = (const float*)d_in[0];
    const float* W        = (const float*)d_in[1];
    const float* a_self   = (const float*)d_in[2];
    const float* a_neigh  = (const float*)d_in[3];
    const float* bias     = (const float*)d_in[4];
    const float* adj_val  = (const float*)d_in[5];
    const int*   edge_row = (const int*)d_in[6];
    const int*   edge_col = (const int*)d_in[7];
    float* out = (float*)d_out;

    // workspace layout (floats)
    float* feats  = (float*)d_ws;                                // H*N*F_OUT
    float* s_self = feats  + (size_t)HEADS * N_NODES * F_OUT;
    float* s_neigh= s_self + (size_t)HEADS * N_NODES;
    float* m      = s_neigh+ (size_t)HEADS * N_NODES;
    float* ssum   = m      + (size_t)HEADS * N_NODES;

    // allow > default dynamic LDS (232448B < 320KB WGP limit); ignore errors
    (void)hipFuncSetAttribute((const void*)gat_proj_wmma_kernel,
                              hipFuncAttributeMaxDynamicSharedMemorySize, SMEM_TOT);

    // 1. projection GEMM (TDM + WMMA)
    dim3 gGemm((N_NODES + 127) / 128, HEADS);
    gat_proj_wmma_kernel<<<gGemm, 256, SMEM_TOT, stream>>>(x, W, feats);

    // 2. score dots
    int nWaves = HEADS * N_NODES;
    gat_scores_kernel<<<(nWaves + 7) / 8, 256, 0, stream>>>(feats, a_self, a_neigh,
                                                            s_self, s_neigh);

    // 3. init accumulators + output
    int initN = N_NODES * HEADS * F_OUT;
    gat_init_kernel<<<(initN + 255) / 256, 256, 0, stream>>>(out, m, ssum);

    // 4. segment max
    int edgeWork = HEADS * N_EDGES;
    gat_segmax_kernel<<<(edgeWork + 255) / 256, 256, 0, stream>>>(
        s_self, s_neigh, adj_val, edge_row, edge_col, m);

    // 5. segment sum of exp
    gat_segsum_kernel<<<(edgeWork + 255) / 256, 256, 0, stream>>>(
        s_self, s_neigh, adj_val, edge_row, edge_col, m, ssum);

    // 6. aggregation scatter
    gat_aggregate_kernel<<<N_EDGES, 256, 0, stream>>>(
        feats, s_self, s_neigh, adj_val, edge_row, edge_col, m, ssum, out);

    // 7. bias + relu
    gat_bias_relu_kernel<<<(initN + 255) / 256, 256, 0, stream>>>(out, bias);
}
